// EthanolSNN_75754633167524
// MI455X (gfx1250) — compile-verified, matrix operand order checked by reading, not run on hardware
//
#include <hip/hip_runtime.h>

// MI455X (gfx1250) SNN inference.
// Memory-bound: 512MB x-stream + 8MB out => ~22us floor @ 23.3 TB/s.
// All LIF state persists in registers across the full T=256 loop; matmuls on
// v_wmma_f32_16x16x32_f16 (A = f16-converted x tile, B = preloaded weights).

typedef __attribute__((ext_vector_type(16))) _Float16 v16h;
typedef __attribute__((ext_vector_type(8)))  _Float16 v8h;
typedef __attribute__((ext_vector_type(8)))  float    v8f;

#define TSTEPS 256
#define BATCH  8192
#define DIN    64
#define H1     28
#define H2     14
#define BETA   0.9f
#define WAVES_PER_BLOCK 4

__device__ __forceinline__ v16h pack_f16(float4 A, float4 B, float4 C, float4 D) {
    v16h r;
    r[0]=(_Float16)A.x; r[1]=(_Float16)A.y; r[2]=(_Float16)A.z; r[3]=(_Float16)A.w;
    r[4]=(_Float16)B.x; r[5]=(_Float16)B.y; r[6]=(_Float16)B.z; r[7]=(_Float16)B.w;
    r[8]=(_Float16)C.x; r[9]=(_Float16)C.y; r[10]=(_Float16)C.z; r[11]=(_Float16)C.w;
    r[12]=(_Float16)D.x; r[13]=(_Float16)D.y; r[14]=(_Float16)D.z; r[15]=(_Float16)D.w;
    return r;
}

__global__ __launch_bounds__(32 * WAVES_PER_BLOCK)
void snn_lif_wmma_kernel(const float* __restrict__ x,
                         const float* __restrict__ W1, const float* __restrict__ b1,
                         const float* __restrict__ W2, const float* __restrict__ b2,
                         const float* __restrict__ W3, const float* __restrict__ b3,
                         float* __restrict__ out)
{
    // Per-wave spike staging buffer for the C-layout -> A-layout pivot.
    __shared__ __align__(16) _Float16 s1_lds[WAVES_PER_BLOCK][16][32];

    const int tid  = threadIdx.x;
    const int wave = tid >> 5;
    const int lane = tid & 31;
    const int mn   = lane & 15;   // row (A-frag) / column (B,C-frag) index
    const int hi   = lane >> 4;   // half-wave selector in WMMA layouts
    const int batch_base = (blockIdx.x * WAVES_PER_BLOCK + wave) * 16;

    // ---- Preload weights into WMMA B-fragment layout (once) ----
    // Half j of a fragment holds K = kc*32 + (j>>3)*16 + hi*8 + (j&7), column N = mn (+tile).
    v16h b1f[2][2];   // [k-chunk][n-tile]
    #pragma unroll
    for (int kc = 0; kc < 2; ++kc) {
        #pragma unroll
        for (int nt = 0; nt < 2; ++nt) {
            #pragma unroll
            for (int j = 0; j < 16; ++j) {
                int k = kc*32 + (j >> 3)*16 + hi*8 + (j & 7);
                int n = nt*16 + mn;
                b1f[kc][nt][j] = (n < H1) ? (_Float16)W1[n*DIN + k] : (_Float16)0.0f;
            }
        }
    }
    v16h b2f;
    #pragma unroll
    for (int j = 0; j < 16; ++j) {
        int k = (j >> 3)*16 + hi*8 + (j & 7);
        int n = mn;
        b2f[j] = (n < H2 && k < H1) ? (_Float16)W2[n*H1 + k] : (_Float16)0.0f;
    }
    const float w3l  = (mn < H2) ? W3[mn] : 0.0f;
    const float b1n0 = b1[mn];                          // n = mn < 16 <= 27
    const float b1n1 = (16 + mn < H1) ? b1[16 + mn] : 0.0f;
    const float b2n  = (mn < H2) ? b2[mn] : 0.0f;
    const float b3s  = b3[0];

    // ---- Persistent LIF membrane state (registers, C-layout) ----
    v8f m1a = {}, m1b = {}, m2v = {};
    float m3r[8] = {};

    // x tile pointer for this lane: row (batch_base+mn), intra-row base hi*8 floats.
    const float4* xp = (const float4*)(x + ((size_t)batch_base + mn) * DIN + hi * 8);
    const size_t xstep4 = (size_t)BATCH * DIN / 4;      // float4 stride per timestep

    for (int t = 0; t < TSTEPS; ++t) {
        // ---- load + convert x tile [16 x 64] fp32 -> two f16 A-fragments ----
        float4 q00 = xp[0], q01 = xp[1], q02 = xp[4],  q03 = xp[5];   // K = 0..7, 16..23 (+hi*8)
        float4 q10 = xp[8], q11 = xp[9], q12 = xp[12], q13 = xp[13];  // K = 32..39, 48..55 (+hi*8)
        if (t + 1 < TSTEPS) __builtin_prefetch(xp + xstep4, 0, 0);    // global_prefetch_b8
        v16h a0 = pack_f16(q00, q01, q02, q03);
        v16h a1 = pack_f16(q10, q11, q12, q13);

        // ---- layer 1: cur1 = x @ W1^T + b1 (4 WMMAs, bias pre-seeded in C) ----
        v8f c0, c1;
        #pragma unroll
        for (int r = 0; r < 8; ++r) { c0[r] = b1n0; c1[r] = b1n1; }
        c0 = __builtin_amdgcn_wmma_f32_16x16x32_f16(false, a0, false, b1f[0][0], (short)0, c0, false, false);
        c0 = __builtin_amdgcn_wmma_f32_16x16x32_f16(false, a1, false, b1f[1][0], (short)0, c0, false, false);
        c1 = __builtin_amdgcn_wmma_f32_16x16x32_f16(false, a0, false, b1f[0][1], (short)0, c1, false, false);
        c1 = __builtin_amdgcn_wmma_f32_16x16x32_f16(false, a1, false, b1f[1][1], (short)0, c1, false, false);

        // ---- LIF layer 1 (subtract reset), spikes -> LDS in [m][k] row-major f16 ----
        #pragma unroll
        for (int r = 0; r < 8; ++r) {
            int m = r + 8 * hi;
            float rst = (m1a[r] > 1.0f) ? 1.0f : 0.0f;
            m1a[r] = BETA * m1a[r] + c0[r] - rst;
            s1_lds[wave][m][mn]      = (_Float16)((m1a[r] > 1.0f) ? 1.0f : 0.0f);
            rst = (m1b[r] > 1.0f) ? 1.0f : 0.0f;
            m1b[r] = BETA * m1b[r] + c1[r] - rst;
            s1_lds[wave][m][16 + mn] = (_Float16)((m1b[r] > 1.0f) ? 1.0f : 0.0f);
        }
        // Same-wave LDS store->load: DS ops are in-order within a wave.

        // ---- read spikes back in A-fragment layout (two 16B LDS reads) ----
        v8h lo = *(const v8h*)&s1_lds[wave][mn][hi * 8];
        v8h hh = *(const v8h*)&s1_lds[wave][mn][16 + hi * 8];
        v16h a2;
        #pragma unroll
        for (int j = 0; j < 8; ++j) { a2[j] = lo[j]; a2[8 + j] = hh[j]; }

        // ---- layer 2: cur2 = s1 @ W2^T + b2 (1 WMMA, K/N zero-padded) ----
        v8f c2;
        #pragma unroll
        for (int r = 0; r < 8; ++r) c2[r] = b2n;
        c2 = __builtin_amdgcn_wmma_f32_16x16x32_f16(false, a2, false, b2f, (short)0, c2, false, false);

        v8f s2v;
        #pragma unroll
        for (int r = 0; r < 8; ++r) {
            float rst = (m2v[r] > 1.0f) ? 1.0f : 0.0f;
            m2v[r] = BETA * m2v[r] + c2[r] - rst;
            s2v[r] = (m2v[r] > 1.0f) ? 1.0f : 0.0f;
        }

        // ---- layer 3 (N=1): cross-lane dot over the 16-lane column dim ----
        float o[8];
        #pragma unroll
        for (int r = 0; r < 8; ++r) {
            float v = s2v[r] * w3l;
            v += __shfl_xor(v, 1);
            v += __shfl_xor(v, 2);
            v += __shfl_xor(v, 4);
            v += __shfl_xor(v, 8);
            float c3 = v + b3s;
            float rst = (m3r[r] > 1.0f) ? 1.0f : 0.0f;
            m3r[r] = BETA * m3r[r] + c3 - rst;
            o[r] = m3r[r];
        }

        // ---- emit mem_rec3[t, batch_base + m] ----
        if (mn == 0) {
            float* op = out + (size_t)t * BATCH + batch_base + 8 * hi;
            float4 o0 = {o[0], o[1], o[2], o[3]};
            float4 o1 = {o[4], o[5], o[6], o[7]};
            ((float4*)op)[0] = o0;
            ((float4*)op)[1] = o1;
        }

        xp += xstep4;
    }
}

extern "C" void kernel_launch(void* const* d_in, const int* in_sizes, int n_in,
                              void* d_out, int out_size, void* d_ws, size_t ws_size,
                              hipStream_t stream) {
    const float* x  = (const float*)d_in[0];
    const float* W1 = (const float*)d_in[1];
    const float* b1 = (const float*)d_in[2];
    const float* W2 = (const float*)d_in[3];
    const float* b2 = (const float*)d_in[4];
    const float* W3 = (const float*)d_in[5];
    const float* b3 = (const float*)d_in[6];
    float* out = (float*)d_out;

    dim3 grid(BATCH / (16 * WAVES_PER_BLOCK));   // 128 blocks
    dim3 block(32 * WAVES_PER_BLOCK);            // 4 waves/block (wave32)
    hipLaunchKernelGGL(snn_lif_wmma_kernel, grid, block, 0, stream,
                       x, W1, b1, W2, b2, W3, b3, out);
}
